// NET_RNN_22780506538384
// MI455X (gfx1250) — matrix operand order repaired and checked by reading, not compiled, not run on hardware
//
#include <hip/hip_runtime.h>
#include <hip/hip_bf16.h>

// ---------------------------------------------------------------------------
// 2-layer LSTM LM for MI455X (gfx1250, wave32, WMMA).
// All matmuls: v_wmma_f32_16x16x32_bf16. H=300 padded to 320.
// Recurrence: persistent cooperative kernel, 1 grid barrier / timestep.
//   - Whh register-resident for the whole sequence (320 VGPRs/wave)
//   - c-state resident in VGPRs
//   - per-step: prefetch xg + ALL 10 A fragments (sched_barrier enforced),
//     then 40 back-to-back WMMAs -> one L2 round-trip per step
//   - 40 WGs x 128 threads: 1 wave/SIMD so ~520 VGPRs/wave always fits
// ---------------------------------------------------------------------------

typedef __bf16 bf16_t;
typedef __attribute__((ext_vector_type(16))) __bf16 v16bf;
typedef __attribute__((ext_vector_type(8)))  __bf16 v8bf;
typedef __attribute__((ext_vector_type(8)))  float  v8f;

#define HDIM   300
#define HP     320
#define KT     (HP/32)       // 10 K-tiles of 32
#define BATCH  128
#define TLEN   512
#define VOCAB  2514
#define VP     2560          // vocab padded to multiple of 64
#define G4     (4*HP)        // 1280 gate columns (padded)
#define MROWS  (BATCH*TLEN)  // 65536
#define SCAN_BLOCKS 40
#define SCAN_TPB    128      // 4 waves -> 160 waves total, 1 quad each

__device__ __forceinline__ v8f wmma_bf16(v16bf a, v16bf b, v8f c) {
  return __builtin_amdgcn_wmma_f32_16x16x32_bf16(false, a, false, b,
                                                 (short)0, c, false, false);
}

// A fragment (16x32, MxK, bf16), row-major source. Lane L<16 holds row L,
// K = k0..k0+7 and k0+16..k0+23 ; lane L>=16 holds row L-16, K shifted by 8.
__device__ __forceinline__ v16bf load_a_frag(const bf16_t* __restrict__ rows,
                                             int stride, int k0, int lane) {
  const bf16_t* p = rows + (size_t)(lane & 15) * stride + k0 + ((lane >> 4) << 3);
  union { v16bf v; v8bf h[2]; } u;
  u.h[0] = *(const v8bf*)(p);
  u.h[1] = *(const v8bf*)(p + 16);
  return u.v;
}

// B fragment (32x16, KxN, bf16) built from W stored [N, Kp] row-major
// (so W row n supplies column n of B = W^T). Lane n<16: K=k0..k0+7 and
// k0+16..k0+23 ; lane n>=16: K shifted by 8 (same order as A fragment).
__device__ __forceinline__ v16bf load_b_frag(const bf16_t* __restrict__ wrows,
                                             int stride, int k0, int lane) {
  const bf16_t* p = wrows + (size_t)(lane & 15) * stride + k0 + ((lane >> 4) << 3);
  union { v16bf v; v8bf h[2]; } u;
  u.h[0] = *(const v8bf*)(p);
  u.h[1] = *(const v8bf*)(p + 16);
  return u.v;
}

__device__ __forceinline__ float sigm(float x) { return 1.0f / (1.0f + __expf(-x)); }
// branch-free tanh: 1 - 2/(e^{2x}+1); saturates correctly for large |x|
__device__ __forceinline__ float tanh_fast(float x) {
  float e = __expf(2.0f * x);
  return 1.0f - 2.0f / (e + 1.0f);
}

// ---------------------------------------------------------------------------
// Weight / bias padding:  src f32 [G, rows, K] -> dst bf16 [G, rows_pad, Kp]
// ---------------------------------------------------------------------------
__global__ void pad_w_kernel(const float* __restrict__ src, bf16_t* __restrict__ dst,
                             int G, int rows, int rows_pad, int K, int Kp) {
  size_t i = (size_t)blockIdx.x * blockDim.x + threadIdx.x;
  size_t tot = (size_t)G * rows_pad * Kp;
  if (i >= tot) return;
  int k = (int)(i % Kp);
  size_t r = i / Kp;
  int g = (int)(r / rows_pad);
  int j = (int)(r % rows_pad);
  float v = 0.f;
  if (j < rows && k < K) v = src[((size_t)g * rows + j) * K + k];
  dst[i] = (bf16_t)v;
}

__global__ void pad_b_kernel(const float* __restrict__ b1, const float* __restrict__ b2,
                             float* __restrict__ dst, int G, int rows, int rows_pad) {
  int i = blockIdx.x * blockDim.x + threadIdx.x;
  if (i >= G * rows_pad) return;
  int g = i / rows_pad, j = i % rows_pad;
  float v = 0.f;
  if (j < rows) { v = b1[g * rows + j]; if (b2) v += b2[g * rows + j]; }
  dst[i] = v;
}

// ---------------------------------------------------------------------------
// Embedding gather -> bf16 padded [M, HP]
// ---------------------------------------------------------------------------
__global__ void embed_kernel(const int* __restrict__ x, const float* __restrict__ emb,
                             bf16_t* __restrict__ out) {
  size_t i = (size_t)blockIdx.x * blockDim.x + threadIdx.x;
  if (i >= (size_t)MROWS * HP) return;
  int col = (int)(i % HP);
  size_t row = i / HP;
  float v = (col < HDIM) ? emb[(size_t)x[row] * HDIM + col] : 0.f;
  out[i] = (bf16_t)v;
}

// ---------------------------------------------------------------------------
// GEMM: C[m,n] = sum_k A[m,k] * W[n,k] + bias[n]
// A bf16 [M,Kp], W bf16 [Npad,Kp]. Workgroup = 8 waves, tile 128(M) x 64(N).
// F32OUT: store f32 with row stride Nreal (masked); else bf16 stride Npad.
// ---------------------------------------------------------------------------
template <bool F32OUT>
__global__ void __launch_bounds__(256) gemm_kernel(
    const bf16_t* __restrict__ A, const bf16_t* __restrict__ W,
    const float* __restrict__ bias, void* __restrict__ outp,
    int Npad, int Nreal, int Kp) {
  const int lane  = threadIdx.x & 31;
  const int wave  = threadIdx.x >> 5;
  const int mbase = blockIdx.x * 128 + wave * 16;
  const int nbase = blockIdx.y * 64;

  const bf16_t* Arow = A + (size_t)mbase * Kp;
  v8f acc[4] = {};
  for (int k0 = 0; k0 < Kp; k0 += 32) {
    v16bf a = load_a_frag(Arow, Kp, k0, lane);
#pragma unroll
    for (int j = 0; j < 4; ++j) {
      const bf16_t* Wrow = W + (size_t)(nbase + 16 * j) * Kp;
      v16bf b = load_b_frag(Wrow, Kp, k0, lane);
      acc[j] = wmma_bf16(a, b, acc[j]);
    }
  }

  const int rowoff = (lane >> 4) * 8;
  const int ncol   = lane & 15;
#pragma unroll
  for (int j = 0; j < 4; ++j) {
    int n = nbase + 16 * j + ncol;
    float bv = bias[n];
#pragma unroll
    for (int r = 0; r < 8; ++r) {
      int m = mbase + r + rowoff;
      float v = acc[j][r] + bv;
      if (F32OUT) {
        if (n < Nreal) ((float*)outp)[(size_t)m * Nreal + n] = v;
      } else {
        ((bf16_t*)outp)[(size_t)m * Npad + n] = (bf16_t)v;
      }
    }
  }
}

// ---------------------------------------------------------------------------
// Scan init: zero ping-pong h buffers and the grid-barrier counter.
// ---------------------------------------------------------------------------
__global__ void scan_init_kernel(bf16_t* __restrict__ hbuf, unsigned* __restrict__ cnt) {
  int i = blockIdx.x * blockDim.x + threadIdx.x;
  if (i == 0) *cnt = 0u;
  if (i < 2 * BATCH * HP) hbuf[i] = (bf16_t)0.f;
}

// ---------------------------------------------------------------------------
// Persistent LSTM scan. Grid must be exactly 40 blocks x 128 threads
// (160 waves). Wave w -> (mt = w/20 batch tile, jt = w%20 hidden-col tile);
// it owns all 4 gate tiles for those columns, so i/f/g/o + c-update happen
// entirely in registers. One grid barrier per timestep (ping-pong h).
// ---------------------------------------------------------------------------
__global__ void __launch_bounds__(SCAN_TPB, 1) lstm_scan_kernel(
    const bf16_t* __restrict__ xg,    // [M, G4] precomputed x-projection + biases
    const bf16_t* __restrict__ Whh,   // [G4, HP] padded, row n = weights of gate-col n
    bf16_t* __restrict__ hbuf,        // [2][BATCH][HP], zeroed
    bf16_t* __restrict__ actOut,      // [M, HP] layer output (bf16)
    unsigned* __restrict__ barrier_cnt) {
  const int lane = threadIdx.x & 31;
  const int gw   = blockIdx.x * (blockDim.x >> 5) + (threadIdx.x >> 5);
  const int mt   = gw / 20;                // batch tile 0..7
  const int jt   = gw % 20;                // hidden-col tile 0..19
  const int rowoff = (lane >> 4) * 8;
  const int j = jt * 16 + (lane & 15);     // hidden column 0..319

  // ---- Whh fragments: loaded once, register-resident for all timesteps ----
  v16bf bfrag[4][KT];
#pragma unroll
  for (int g = 0; g < 4; ++g) {
    const bf16_t* Wrow = Whh + (size_t)(g * HP + jt * 16) * HP;
#pragma unroll
    for (int kk = 0; kk < KT; ++kk)
      bfrag[g][kk] = load_b_frag(Wrow, HP, kk * 32, lane);
  }

  v8f cstate = {};
  unsigned expected = 0;
  const unsigned nb = gridDim.x;

  for (int t = 0; t < TLEN; ++t) {
    const bf16_t* hin  = hbuf + (size_t)(t & 1) * (BATCH * HP);
    bf16_t*       hout = hbuf + (size_t)((t + 1) & 1) * (BATCH * HP);

    // ---- prefetch xg for this step (independent of the matmul) ----
    float xv[4][8];
#pragma unroll
    for (int r = 0; r < 8; ++r) {
      int b = mt * 16 + r + rowoff;
      size_t xrow = ((size_t)b * TLEN + t) * G4;
#pragma unroll
      for (int g = 0; g < 4; ++g)
        xv[g][r] = (float)xg[xrow + g * HP + j];
    }

    // ---- prefetch the FULL A operand (h_{t-1} rows of this batch tile) ----
    const bf16_t* Arow = hin + (size_t)(mt * 16) * HP;
    v16bf afrag[KT];
#pragma unroll
    for (int kk = 0; kk < KT; ++kk)
      afrag[kk] = load_a_frag(Arow, HP, kk * 32, lane);

    // Hard scheduling fence: all loads above must be issued before any WMMA
    // below (prevents the scheduler from collapsing afrag[] into a rotating
    // buffer with one s_wait_loadcnt per K-tile -> one L2 round-trip/step).
    __builtin_amdgcn_sched_barrier(0);

    // ---- 40 back-to-back WMMAs: gates = h_{t-1} @ Whh^T ----
    v8f acc[4] = {};
#pragma unroll
    for (int kk = 0; kk < KT; ++kk) {
#pragma unroll
      for (int g = 0; g < 4; ++g)
        acc[g] = wmma_bf16(afrag[kk], bfrag[g][kk], acc[g]);
    }

    // ---- LSTM cell elementwise, c kept in VGPRs for the whole sequence ----
#pragma unroll
    for (int r = 0; r < 8; ++r) {
      int b = mt * 16 + r + rowoff;                  // batch row
      float iv = acc[0][r] + xv[0][r];
      float fv = acc[1][r] + xv[1][r];
      float gv = acc[2][r] + xv[2][r];
      float ov = acc[3][r] + xv[3][r];
      float c  = sigm(fv) * cstate[r] + sigm(iv) * tanh_fast(gv);
      cstate[r] = c;
      float hv = sigm(ov) * tanh_fast(c);
      bf16_t hb = (bf16_t)hv;
      hout[(size_t)b * HP + j] = hb;
      actOut[((size_t)b * TLEN + t) * HP + j] = hb;
    }

    // ---- grid-wide barrier: all h_t stores visible before t+1 reads ----
    expected += nb;
    __syncthreads();
    if (threadIdx.x == 0) {
      __threadfence();
      atomicAdd(barrier_cnt, 1u);
      while (__hip_atomic_load(barrier_cnt, __ATOMIC_RELAXED,
                               __HIP_MEMORY_SCOPE_AGENT) < expected) {
        __builtin_amdgcn_s_sleep(1);
      }
    }
    __syncthreads();
    __threadfence();
  }
}

// ---------------------------------------------------------------------------
// Host side
// ---------------------------------------------------------------------------
extern "C" void kernel_launch(void* const* d_in, const int* in_sizes, int n_in,
                              void* d_out, int out_size, void* d_ws, size_t ws_size,
                              hipStream_t stream) {
  const int*   x    = (const int*)  d_in[0];
  const float* emb  = (const float*)d_in[1];
  const float* Wih0 = (const float*)d_in[2];
  const float* Whh0 = (const float*)d_in[3];
  const float* bih0 = (const float*)d_in[4];
  const float* bhh0 = (const float*)d_in[5];
  const float* Wih1 = (const float*)d_in[6];
  const float* Whh1 = (const float*)d_in[7];
  const float* bih1 = (const float*)d_in[8];
  const float* bhh1 = (const float*)d_in[9];
  const float* Wout = (const float*)d_in[10];
  const float* bout = (const float*)d_in[11];
  float* out = (float*)d_out;

  char* ws = (char*)d_ws;
  size_t off = 0;
  auto alloc = [&](size_t bytes) {
    size_t o = off;
    off = (off + bytes + 255) & ~(size_t)255;
    return (void*)(ws + o);
  };
  bf16_t* Wih0p = (bf16_t*)alloc((size_t)G4 * HP * 2);
  bf16_t* Whh0p = (bf16_t*)alloc((size_t)G4 * HP * 2);
  bf16_t* Wih1p = (bf16_t*)alloc((size_t)G4 * HP * 2);
  bf16_t* Whh1p = (bf16_t*)alloc((size_t)G4 * HP * 2);
  bf16_t* Woutp = (bf16_t*)alloc((size_t)VP * HP * 2);
  float*  b0p   = (float*) alloc((size_t)G4 * 4);
  float*  b1p   = (float*) alloc((size_t)G4 * 4);
  float*  boutp = (float*) alloc((size_t)VP * 4);
  bf16_t* hbuf  = (bf16_t*)alloc((size_t)2 * BATCH * HP * 2);
  unsigned* cnt = (unsigned*)alloc(256);
  bf16_t* actA  = (bf16_t*)alloc((size_t)MROWS * HP * 2);   // e, then layer-2 output
  bf16_t* actB  = (bf16_t*)alloc((size_t)MROWS * HP * 2);   // layer-1 output
  bf16_t* xg    = (bf16_t*)alloc((size_t)MROWS * G4 * 2);   // gate pre-activations
  (void)ws_size; (void)n_in; (void)in_sizes; (void)out_size;

  const int TPB = 256;
  auto blks = [](size_t n, int t) { return (unsigned)((n + t - 1) / t); };

  // --- pad weights (gate-major [4,300,300] -> [4,320,320]) & biases ---
  size_t wgate = (size_t)4 * HP * HP;
  pad_w_kernel<<<blks(wgate, TPB), TPB, 0, stream>>>(Wih0, Wih0p, 4, HDIM, HP, HDIM, HP);
  pad_w_kernel<<<blks(wgate, TPB), TPB, 0, stream>>>(Whh0, Whh0p, 4, HDIM, HP, HDIM, HP);
  pad_w_kernel<<<blks(wgate, TPB), TPB, 0, stream>>>(Wih1, Wih1p, 4, HDIM, HP, HDIM, HP);
  pad_w_kernel<<<blks(wgate, TPB), TPB, 0, stream>>>(Whh1, Whh1p, 4, HDIM, HP, HDIM, HP);
  pad_w_kernel<<<blks((size_t)VP * HP, TPB), TPB, 0, stream>>>(Wout, Woutp, 1, VOCAB, VP, HDIM, HP);
  pad_b_kernel<<<blks(G4, TPB), TPB, 0, stream>>>(bih0, bhh0, b0p, 4, HDIM, HP);
  pad_b_kernel<<<blks(G4, TPB), TPB, 0, stream>>>(bih1, bhh1, b1p, 4, HDIM, HP);
  pad_b_kernel<<<blks(VP, TPB), TPB, 0, stream>>>(bout, nullptr, boutp, 1, VOCAB, VP);

  // --- embedding gather ---
  embed_kernel<<<blks((size_t)MROWS * HP, TPB), TPB, 0, stream>>>(x, emb, actA);

  dim3 gXg(MROWS / 128, G4 / 64);     // 512 x 20
  dim3 gOut(MROWS / 128, VP / 64);    // 512 x 40

  // --- layer 0: input projection, then scan ---
  gemm_kernel<false><<<gXg, TPB, 0, stream>>>(actA, Wih0p, b0p, xg, G4, G4, HP);
  scan_init_kernel<<<blks(2 * BATCH * HP, TPB), TPB, 0, stream>>>(hbuf, cnt);
  lstm_scan_kernel<<<SCAN_BLOCKS, SCAN_TPB, 0, stream>>>(xg, Whh0p, hbuf, actB, cnt);

  // --- layer 1 ---
  gemm_kernel<false><<<gXg, TPB, 0, stream>>>(actB, Wih1p, b1p, xg, G4, G4, HP);
  scan_init_kernel<<<blks(2 * BATCH * HP, TPB), TPB, 0, stream>>>(hbuf, cnt);
  lstm_scan_kernel<<<SCAN_BLOCKS, SCAN_TPB, 0, stream>>>(xg, Whh1p, hbuf, actA, cnt);

  // --- output projection: logits f32 straight to d_out ---
  gemm_kernel<true><<<gOut, TPB, 0, stream>>>(actA, Woutp, boutp, out, VP, VOCAB, HP);
}
